// KGLinkPredictor_58660663329127
// MI455X (gfx1250) — matrix operand order repaired and checked by reading, not compiled, not run on hardware
//
#include <hip/hip_runtime.h>

// ---------------------------------------------------------------------------
// KG link predictor on MI455X (gfx1250, wave32, WMMA)
//   GAT layer:  h = X@W (WMMA f16->f32), per-edge attention softmax via
//               L2 atomics, aggregation via global_atomic_add_f32.
//   Scan:       column-private sequential update (no barriers needed).
//   Decode:     DistMult + sigmoid, one wave per triplet.
// ---------------------------------------------------------------------------

typedef __attribute__((ext_vector_type(16))) _Float16 v16h;
typedef __attribute__((ext_vector_type(8)))  float    v8f;

constexpr int   NN  = 50000;
constexpr int   EE  = 800000;
constexpr int   HH  = 256;
constexpr int   BB  = 256;
constexpr int   KK  = 16;
constexpr int   ERR = 20000;
constexpr float LAM = 0.7f;
constexpr int   MT  = NN / 16;   // 3125 row tiles

// ---------------- operand swizzle into CDNA5 WMMA lane layout ---------------
// A (16x32 f16): lane L holds row M=L&15; K pattern kb=(L<16?0:8):
//   elems 0..7 -> K = kb+e ; elems 8..15 -> K = 16+kb+(e-8)
__global__ void k_swizzle_a(const float* __restrict__ src, _Float16* __restrict__ dst) {
  int tid  = blockIdx.x * 256 + threadIdx.x;          // MT*8*32*16 = 12.8M
  int e    = tid & 15;
  int lane = (tid >> 4) & 31;
  int kt   = (tid >> 9) & 7;
  int rt   = tid >> 12;
  int row  = rt * 16 + (lane & 15);
  int kb   = (lane & 16) ? 8 : 0;
  int kk   = kt * 32 + (e < 8 ? kb + e : 16 + kb + (e - 8));
  dst[tid] = (_Float16)src[(size_t)row * HH + kk];
}

// B (32x16 f16, K x N): lane L holds col N=L&15; kb=(L<16?0:8):
//   elems 0..7 -> K = kb+e ; elems 8..15 -> K = 16+kb+(e-8)
__global__ void k_swizzle_w(const float* __restrict__ W, _Float16* __restrict__ dst) {
  int tid  = blockIdx.x * 256 + threadIdx.x;          // 16*8*32*16 = 65536
  int e    = tid & 15;
  int lane = (tid >> 4) & 31;
  int kt   = (tid >> 9) & 7;
  int ct   = tid >> 12;
  int col  = ct * 16 + (lane & 15);
  int koff = (lane & 16) ? 8 : 0;
  int kk   = kt * 32 + (e < 8 ? koff + e : 16 + koff + (e - 8));
  dst[tid] = (_Float16)W[(size_t)kk * HH + col];
}

// ------------------------- WMMA GEMM  C = A @ W ----------------------------
// grid.x = MT row tiles, 256 threads = 8 waves, wave w -> column tiles {2w,2w+1}
__global__ void k_gemm(const _Float16* __restrict__ Aswz,
                       const _Float16* __restrict__ Wswz,
                       float* __restrict__ C) {
  int wave = threadIdx.x >> 5;
  int lane = threadIdx.x & 31;
  int rt   = blockIdx.x;
  const v16h* A = (const v16h*)Aswz;
  const v16h* B = (const v16h*)Wswz;
  int ct0 = wave * 2, ct1 = wave * 2 + 1;
  v8f acc0 = {}, acc1 = {};
#pragma unroll
  for (int kt = 0; kt < 8; ++kt) {
    v16h a  = A[(rt  * 8 + kt) * 32 + lane];
    v16h b0 = B[(ct0 * 8 + kt) * 32 + lane];
    v16h b1 = B[(ct1 * 8 + kt) * 32 + lane];
    acc0 = __builtin_amdgcn_wmma_f32_16x16x32_f16(false, a, false, b0, (short)0, acc0, false, false);
    acc1 = __builtin_amdgcn_wmma_f32_16x16x32_f16(false, a, false, b1, (short)0, acc1, false, false);
  }
  int n       = lane & 15;
  int rowbase = rt * 16 + ((lane & 16) ? 8 : 0);
#pragma unroll
  for (int v = 0; v < 8; ++v) {
    C[(size_t)(rowbase + v) * HH + ct0 * 16 + n] = acc0[v];
    C[(size_t)(rowbase + v) * HH + ct1 * 16 + n] = acc1[v];
  }
}

// ---------------- per-node attention dots: hs = h@a_src, hd = h@a_dst ------
__global__ void k_rowdot(const float* __restrict__ h,
                         const float* __restrict__ as_, const float* __restrict__ ad_,
                         float* __restrict__ hs, float* __restrict__ hd) {
  int row  = blockIdx.x * 8 + (threadIdx.x >> 5);
  int lane = threadIdx.x & 31;
  float s1 = 0.f, s2 = 0.f;
  const float* hr = h + (size_t)row * HH;
#pragma unroll
  for (int d = 0; d < 8; ++d) {
    float v = hr[lane + 32 * d];
    s1 += v * as_[lane + 32 * d];
    s2 += v * ad_[lane + 32 * d];
  }
#pragma unroll
  for (int off = 16; off > 0; off >>= 1) {
    s1 += __shfl_down(s1, off, 32);
    s2 += __shfl_down(s2, off, 32);
  }
  if (lane == 0) { hs[row] = s1; hd[row] = s2; }
}

// ------------------- edge pass 1: leaky-relu + segment max -----------------
__device__ __forceinline__ unsigned f32_ordered(float f) {
  unsigned u = __float_as_uint(f);
  return (u & 0x80000000u) ? ~u : (u | 0x80000000u);
}

__global__ void k_edge_max(const float* __restrict__ hs, const float* __restrict__ hd,
                           const int* __restrict__ ei,
                           float* __restrict__ eval, unsigned* __restrict__ mkey) {
  int e = blockIdx.x * 256 + threadIdx.x;
  int s = ei[e], d = ei[EE + e];
  float sv = hs[s] + hd[d];
  float lr = sv > 0.f ? sv : 0.2f * sv;
  eval[e] = lr;
  atomicMax(&mkey[d], f32_ordered(lr));
}

// decode ordered-uint max; untouched (0) segments -> 0.0 per reference
__global__ void k_fix_m(const unsigned* __restrict__ mkey, float* __restrict__ m) {
  int n = blockIdx.x * 256 + threadIdx.x;
  if (n >= NN) return;
  unsigned key = mkey[n];
  float f;
  if (key == 0u)                  f = 0.f;
  else if (key & 0x80000000u)     f = __uint_as_float(key & 0x7fffffffu);
  else                            f = __uint_as_float(~key);
  m[n] = f;
}

// ------------------- edge pass 2: exp + segment sum ------------------------
__global__ void k_edge_exp(const int* __restrict__ ei, const float* __restrict__ m,
                           float* __restrict__ eval, float* __restrict__ z) {
  int e = blockIdx.x * 256 + threadIdx.x;
  int d = ei[EE + e];
  float p = expf(eval[e] - m[d]);
  eval[e] = p;
  atomicAdd(&z[d], p);
}

// ------------------- edge pass 3: weighted aggregation ---------------------
// one wave per edge; lane covers 8 feature dims (coalesced 1KB row reads)
__global__ void k_aggregate(const float* __restrict__ h, const float* __restrict__ eval,
                            const float* __restrict__ z, const int* __restrict__ ei,
                            float* __restrict__ out) {
  int e    = blockIdx.x * 8 + (threadIdx.x >> 5);
  int lane = threadIdx.x & 31;
  int s = ei[e], d = ei[EE + e];
  float alpha = eval[e] / (z[d] + 1e-16f);
  const float4* hr = (const float4*)(h + (size_t)s * HH) + lane * 2;
  float4 v0 = hr[0], v1 = hr[1];
  float* orow = out + (size_t)d * HH + lane * 8;
  atomicAdd(&orow[0], alpha * v0.x); atomicAdd(&orow[1], alpha * v0.y);
  atomicAdd(&orow[2], alpha * v0.z); atomicAdd(&orow[3], alpha * v0.w);
  atomicAdd(&orow[4], alpha * v1.x); atomicAdd(&orow[5], alpha * v1.y);
  atomicAdd(&orow[6], alpha * v1.z); atomicAdd(&orow[7], alpha * v1.w);
}

// ------------------- bias add (+ optional SiLU) ----------------------------
template <bool SILU>
__global__ void k_bias_act(float* __restrict__ buf, const float* __restrict__ bias) {
  size_t i = (size_t)blockIdx.x * 256 + threadIdx.x;
  float c = buf[i] + bias[i & (HH - 1)];
  if (SILU) c = c / (1.f + expf(-c));
  buf[i] = c;
}

// ------------- per-triplet degree/coefficient pre-pass (parallel) ----------
__global__ void k_deg(const int* __restrict__ head, const int* __restrict__ rel,
                      const int* __restrict__ g2l, const int* __restrict__ res,
                      float* __restrict__ cvals) {
  __shared__ int sh[256];
  int b = blockIdx.x;
  int r = rel[b];
  bool dis = (r >= 4) && (r <= 6);
  int ridx  = dis ? (r - 4) : 0;
  int local = g2l[head[b]];
  int cnt = 0;
  for (int j = threadIdx.x; j < ERR; j += 256)
    cnt += (res[ridx * ERR + j] == local) ? 1 : 0;
  sh[threadIdx.x] = cnt;
  __syncthreads();
  for (int off = 128; off > 0; off >>= 1) {
    if (threadIdx.x < off) sh[threadIdx.x] += sh[threadIdx.x + off];
    __syncthreads();
  }
  if (threadIdx.x == 0) {
    float deg = (float)sh[0];
    cvals[b] = dis ? (LAM * expf(-LAM * deg) + 0.2f) : -1.0f;
  }
}

// -------- sequential scan: column-private, one block, no barriers ----------
__global__ void k_scan(float* __restrict__ emb, const int* __restrict__ head,
                       const int* __restrict__ g2l, const int* __restrict__ sn,
                       const float* __restrict__ sw, const float* __restrict__ cvals) {
  int t = threadIdx.x;   // feature column owned exclusively by this thread
  for (int i = 0; i < BB; ++i) {
    float c = cvals[i];
    if (c < 0.f) continue;           // non-disease triplet: row unchanged
    int hi    = head[i];
    int local = g2l[hi];
    float vec = 0.f;
#pragma unroll
    for (int k = 0; k < KK; ++k) {
      int nb = sn[local * KK + k];
      vec += sw[local * KK + k] * emb[(size_t)nb * HH + t];
    }
    float old = emb[(size_t)hi * HH + t];
    emb[(size_t)hi * HH + t] = c * vec + (1.f - c) * old;
  }
}

// --------------------- DistMult decode + sigmoid ---------------------------
__global__ void k_decode(const float* __restrict__ emb, const int* __restrict__ head,
                         const int* __restrict__ tail, const int* __restrict__ rel,
                         const float* __restrict__ rel_emb, float* __restrict__ out) {
  int b    = blockIdx.x * 8 + (threadIdx.x >> 5);
  int lane = threadIdx.x & 31;
  int hi = head[b], ti = tail[b], r = rel[b];
  float s = 0.f;
#pragma unroll
  for (int d = 0; d < 8; ++d) {
    int c = lane + 32 * d;
    s += emb[(size_t)hi * HH + c] * rel_emb[(size_t)r * HH + c] * emb[(size_t)ti * HH + c];
  }
#pragma unroll
  for (int off = 16; off > 0; off >>= 1) s += __shfl_down(s, off, 32);
  if (lane == 0) out[b] = 1.f / (1.f + expf(-s));
}

// ---------------------------------------------------------------------------
extern "C" void kernel_launch(void* const* d_in, const int* in_sizes, int n_in,
                              void* d_out, int out_size, void* d_ws, size_t ws_size,
                              hipStream_t stream) {
  (void)in_sizes; (void)n_in; (void)out_size; (void)ws_size;
  const float* x    = (const float*)d_in[0];
  const int*   ei   = (const int*)  d_in[1];
  const int*   head = (const int*)  d_in[2];
  const int*   rel  = (const int*)  d_in[3];
  const int*   tail = (const int*)  d_in[4];
  const int*   g2l  = (const int*)  d_in[5];
  const int*   res  = (const int*)  d_in[6];
  const int*   sn   = (const int*)  d_in[7];
  const float* sw   = (const float*)d_in[8];
  const float* W1   = (const float*)d_in[9];
  const float* a1s  = (const float*)d_in[10];
  const float* a1d  = (const float*)d_in[11];
  const float* b1   = (const float*)d_in[12];
  const float* W2   = (const float*)d_in[13];
  const float* a2s  = (const float*)d_in[14];
  const float* a2d  = (const float*)d_in[15];
  const float* b2   = (const float*)d_in[16];
  const float* rem  = (const float*)d_in[17];
  float* out = (float*)d_out;

  // workspace carve-out (everything fits comfortably in L2-sized working sets)
  char* base = (char*)d_ws;
  auto carve = [&](size_t bytes) { char* p = base; base += (bytes + 255) & ~(size_t)255; return p; };
  float*     bufA  = (float*)    carve((size_t)NN * HH * 4);   // h (pre-agg features)
  float*     bufB  = (float*)    carve((size_t)NN * HH * 4);   // agg1 / silu
  float*     bufC  = (float*)    carve((size_t)NN * HH * 4);   // node_emb
  _Float16*  swzA  = (_Float16*) carve((size_t)NN * HH * 2);   // WMMA-layout A
  _Float16*  swzW1 = (_Float16*) carve((size_t)HH * HH * 2);
  _Float16*  swzW2 = (_Float16*) carve((size_t)HH * HH * 2);
  float*     hs    = (float*)    carve((size_t)NN * 4);
  float*     hd    = (float*)    carve((size_t)NN * 4);
  unsigned*  mkey  = (unsigned*) carve((size_t)NN * 4);        // reused as float m
  float*     zz    = (float*)    carve((size_t)NN * 4);
  float*     eval  = (float*)    carve((size_t)EE * 4);
  float*     cvals = (float*)    carve((size_t)BB * 4);

  const int swzA_blocks = MT * 8 * 32 * 16 / 256;   // 50000
  const int nh_blocks   = (size_t)NN * HH / 256;    // 50000

  hipMemsetAsync(mkey, 0, (size_t)NN * 4, stream);
  hipMemsetAsync(zz,   0, (size_t)NN * 4, stream);
  hipMemsetAsync(bufB, 0, (size_t)NN * HH * 4, stream);
  hipMemsetAsync(bufC, 0, (size_t)NN * HH * 4, stream);

  // weights -> WMMA B layout (f16)
  k_swizzle_w<<<256, 256, 0, stream>>>(W1, swzW1);
  k_swizzle_w<<<256, 256, 0, stream>>>(W2, swzW2);

  // ---------------- layer 1 ----------------
  k_swizzle_a<<<swzA_blocks, 256, 0, stream>>>(x, swzA);
  k_gemm<<<MT, 256, 0, stream>>>(swzA, swzW1, bufA);
  k_rowdot<<<NN / 8, 256, 0, stream>>>(bufA, a1s, a1d, hs, hd);
  k_edge_max<<<EE / 256, 256, 0, stream>>>(hs, hd, ei, eval, mkey);
  k_fix_m<<<(NN + 255) / 256, 256, 0, stream>>>(mkey, (float*)mkey);
  k_edge_exp<<<EE / 256, 256, 0, stream>>>(ei, (const float*)mkey, eval, zz);
  k_aggregate<<<EE / 8, 256, 0, stream>>>(bufA, eval, zz, ei, bufB);
  k_bias_act<true><<<nh_blocks, 256, 0, stream>>>(bufB, b1);   // + b1, SiLU

  // ---------------- layer 2 ----------------
  hipMemsetAsync(mkey, 0, (size_t)NN * 4, stream);
  hipMemsetAsync(zz,   0, (size_t)NN * 4, stream);
  k_swizzle_a<<<swzA_blocks, 256, 0, stream>>>(bufB, swzA);
  k_gemm<<<MT, 256, 0, stream>>>(swzA, swzW2, bufA);
  k_rowdot<<<NN / 8, 256, 0, stream>>>(bufA, a2s, a2d, hs, hd);
  k_edge_max<<<EE / 256, 256, 0, stream>>>(hs, hd, ei, eval, mkey);
  k_fix_m<<<(NN + 255) / 256, 256, 0, stream>>>(mkey, (float*)mkey);
  k_edge_exp<<<EE / 256, 256, 0, stream>>>(ei, (const float*)mkey, eval, zz);
  k_aggregate<<<EE / 8, 256, 0, stream>>>(bufA, eval, zz, ei, bufC);
  k_bias_act<false><<<nh_blocks, 256, 0, stream>>>(bufC, b2);  // + b2 (node_emb)

  // ---------------- scan + decode ----------------
  k_deg<<<BB, 256, 0, stream>>>(head, rel, g2l, res, cvals);
  k_scan<<<1, 256, 0, stream>>>(bufC, head, g2l, sn, sw, cvals);
  k_decode<<<BB / 8, 256, 0, stream>>>(bufC, head, tail, rel, rem, out);
}
